// MllamaTextCrossAttention_40484361732229
// MI455X (gfx1250) — compile-verified, matrix operand order checked by reading, not run on hardware
//
#include <hip/hip_runtime.h>

#define HD    128
#define NH    32
#define HID   4096
#define QLEN  1024
#define KVLEN 4096

typedef __bf16 bf16;
typedef __bf16 bf16x8  __attribute__((ext_vector_type(8)));
typedef __bf16 bf16x16 __attribute__((ext_vector_type(16)));
typedef float  f32x4   __attribute__((ext_vector_type(4)));
typedef float  v8f     __attribute__((ext_vector_type(8)));
typedef int    v4i     __attribute__((ext_vector_type(4)));

// ---- async global->LDS copy (gfx1250 GLOBAL_LOAD_ASYNC_TO_LDS_B128).
// Probe result: builtin exists here with signature
//   (addrspace(1) v4i*, addrspace(3) v4i*, int offset, int cpol)
#if defined(__has_builtin)
#if __has_builtin(__builtin_amdgcn_global_load_async_to_lds_b128)
#define ASYNC_LDS 1
#endif
#endif
#ifndef ASYNC_LDS
#define ASYNC_LDS 0
#endif

__device__ inline void cp16_g2l(const bf16* g, bf16* l) {
#if ASYNC_LDS
  __builtin_amdgcn_global_load_async_to_lds_b128(
      (__attribute__((address_space(1))) v4i*)(void*)g,
      (__attribute__((address_space(3))) v4i*)l, 0, 0);
#else
  *(bf16x8*)l = *(const bf16x8*)g;
#endif
}

__device__ inline void async_wait0() {
#if ASYNC_LDS
#if __has_builtin(__builtin_amdgcn_s_wait_asynccnt)
  __builtin_amdgcn_s_wait_asynccnt(0);
#else
  asm volatile("s_wait_asynccnt 0" ::: "memory");
#endif
#endif
}

__device__ inline v8f zero8() {
  v8f z;
#pragma unroll
  for (int i = 0; i < 8; ++i) z[i] = 0.0f;
  return z;
}

// D = A*B + C,  A 16x32 bf16, B 32x16 bf16, C/D 16x16 f32
__device__ inline v8f wmma_bf16(bf16x16 a, bf16x16 b, v8f c) {
  return __builtin_amdgcn_wmma_f32_16x16x32_bf16(false, a, false, b,
                                                 (short)0, c, false, false);
}

// WMMA operand fragment: lane's K chunk is [hi*8..+7] and [16+hi*8..+7].
__device__ inline bf16x16 ld_frag(const bf16* p, int hi) {
  bf16x8 lo = *(const bf16x8*)(p + hi * 8);
  bf16x8 hb = *(const bf16x8*)(p + 16 + hi * 8);
  bf16x16 r;
#pragma unroll
  for (int i = 0; i < 8; ++i) { r[i] = lo[i]; r[8 + i] = hb[i]; }
  return r;
}

// fp32x8 -> bf16x8 convert for LDS staging
__device__ inline bf16x8 cvt8(const float* src) {
  f32x4 f0 = *(const f32x4*)src;
  f32x4 f1 = *(const f32x4*)(src + 4);
  bf16x8 v;
#pragma unroll
  for (int j = 0; j < 4; ++j) { v[j] = (bf16)f0[j]; v[4 + j] = (bf16)f1[j]; }
  return v;
}

// ---------------------------------------------------------------------------
// C[M,N] = A[M,K] @ B[N,K]^T  (nn.Linear x @ W^T).
// Block tile 128(M) x 256(N), 256 threads = 8 waves as 2(M) x 4(N);
// wave tile 64x64 = 4x4 WMMA accumulators (16 WMMA per 32-K step).
// Double-buffered LDS; bf16 A staged via async LDS DMA when available.
// ---------------------------------------------------------------------------
template <bool A_BF16, bool C_BF16>
__global__ __launch_bounds__(256) void gemm_xwt(const void* __restrict__ Ap,
                                                const float* __restrict__ Bp,
                                                void* __restrict__ Cp,
                                                int M, int N, int K) {
  __shared__ bf16 sA[2][128 * 32];   // 2 x 8KB
  __shared__ bf16 sB[2][256 * 32];   // 2 x 16KB
  const int nBlock = blockIdx.x * 256;
  const int mBlock = blockIdx.y * 128;
  const int tid  = threadIdx.x;
  const int wave = tid >> 5;
  const int lane = tid & 31;
  const int lm   = lane & 15;
  const int hi   = lane >> 4;
  const int wm   = wave & 1;   // 64-row slice of 128
  const int wn   = wave >> 1;  // 64-col slice of 256

  v8f acc[4][4];
#pragma unroll
  for (int i = 0; i < 4; ++i)
#pragma unroll
    for (int j = 0; j < 4; ++j) acc[i][j] = zero8();

  const int ksteps = K >> 5;

  auto load_tile = [&](int kb, int buf) {
    const int k0 = kb << 5;
#pragma unroll
    for (int i = 0; i < 2; ++i) {  // A: 128 rows x 4 chunks of 8
      int task = tid + i * 256;
      int row  = task >> 2;
      int c8   = (task & 3) << 3;
      if (A_BF16) {
        cp16_g2l((const bf16*)Ap + (size_t)(mBlock + row) * K + k0 + c8,
                 &sA[buf][row * 32 + c8]);
      } else {
        const float* src = (const float*)Ap + (size_t)(mBlock + row) * K + k0 + c8;
        *(bf16x8*)&sA[buf][row * 32 + c8] = cvt8(src);
      }
    }
#pragma unroll
    for (int i = 0; i < 4; ++i) {  // B: 256 rows x 4 chunks of 8
      int task = tid + i * 256;
      int row  = task >> 2;
      int c8   = (task & 3) << 3;
      const float* src = Bp + (size_t)(nBlock + row) * K + k0 + c8;
      *(bf16x8*)&sB[buf][row * 32 + c8] = cvt8(src);
      if (kb + 1 < ksteps) __builtin_prefetch(src + 32, 0, 0);  // global_prefetch_b8
    }
  };

  load_tile(0, 0);
  int cur = 0;
  for (int kb = 0; kb < ksteps; ++kb) {
    if (A_BF16) async_wait0();   // tile kb's async A copies (ours) complete
    __syncthreads();             // all waves' stages visible; old buffer free
    if (kb + 1 < ksteps) load_tile(kb + 1, cur ^ 1);

    bf16x16 bfr[4];
#pragma unroll
    for (int nj = 0; nj < 4; ++nj)
      bfr[nj] = ld_frag(&sB[cur][(wn * 64 + nj * 16 + lm) * 32], hi);
#pragma unroll
    for (int mi = 0; mi < 4; ++mi) {
      bf16x16 afr = ld_frag(&sA[cur][(wm * 64 + mi * 16 + lm) * 32], hi);
#pragma unroll
      for (int nj = 0; nj < 4; ++nj)
        acc[mi][nj] = wmma_bf16(afr, bfr[nj], acc[mi][nj]);
    }
    cur ^= 1;
  }

#pragma unroll
  for (int mi = 0; mi < 4; ++mi)
#pragma unroll
    for (int nj = 0; nj < 4; ++nj)
#pragma unroll
      for (int r = 0; r < 8; ++r) {
        int row = mBlock + wm * 64 + mi * 16 + r + 8 * hi;
        int col = nBlock + wn * 64 + nj * 16 + lm;
        if (C_BF16) ((bf16*)Cp)[(size_t)row * N + col] = (bf16)acc[mi][nj][r];
        else        ((float*)Cp)[(size_t)row * N + col] = acc[mi][nj][r];
      }
}

// ---------------------------------------------------------------------------
// Per-head RMSNorm over d=128 (fp32 math) -> bf16; one wave per (row, head).
// ---------------------------------------------------------------------------
__global__ __launch_bounds__(256) void rmsnorm_kernel(const float* __restrict__ X,
                                                      const float* __restrict__ w,
                                                      bf16* __restrict__ out,
                                                      float scale) {
  int g    = blockIdx.x * 8 + (threadIdx.x >> 5);
  int lane = threadIdx.x & 31;
  int row  = g >> 5;
  int head = g & 31;
  size_t base = (size_t)row * HID + head * HD;
  int i0 = lane * 4;
  f32x4 x = *(const f32x4*)(X + base + i0);
  float ss = x[0] * x[0] + x[1] * x[1] + x[2] * x[2] + x[3] * x[3];
#pragma unroll
  for (int off = 16; off; off >>= 1) ss += __shfl_xor(ss, off, 32);
  float r = rsqrtf(ss * (1.0f / 128.0f) + 1e-5f) * scale;
  f32x4 wv = *(const f32x4*)(w + i0);
#pragma unroll
  for (int j = 0; j < 4; ++j) out[base + i0 + j] = (bf16)(x[j] * r * wv[j]);
}

// ---------------------------------------------------------------------------
// Flash attention: block = (head, 128 q-rows), 8 waves x 16 rows.
// K chunk staged via async LDS DMA; V staged transposed (bf16 scalar stores);
// online fp32 softmax with 16-lane shfl_xor row reductions (matches the
// WMMA C/D layout: VGPR r <-> row r+8*(lane>=16), columns in lanes);
// P tile aliases the K region after a barrier.
// ---------------------------------------------------------------------------
__global__ __launch_bounds__(256) void flash_attn_kernel(const bf16* __restrict__ Q,
                                                         const bf16* __restrict__ Km,
                                                         const bf16* __restrict__ V,
                                                         bf16* __restrict__ O) {
  __shared__ bf16 sK[128 * 128];    // K chunk [kv][d]
  __shared__ bf16 sVT[128 * 128];   // V chunk transposed [d][kv]
  const int tid  = threadIdx.x;
  const int wave = tid >> 5;
  const int lane = tid & 31;
  const int lm   = lane & 15;
  const int hi   = lane >> 4;
  const int h    = blockIdx.y;
  const int q0   = blockIdx.x * 128 + wave * 16;
  bf16* sP = sK + wave * (16 * 128);  // per-wave P slice aliases K region

  bf16x16 qf[4];
#pragma unroll
  for (int kf = 0; kf < 4; ++kf)
    qf[kf] = ld_frag(Q + (size_t)(q0 + lm) * HID + h * HD + kf * 32, hi);

  v8f o[8];
  float mrow[8], lrow[8];
#pragma unroll
  for (int t = 0; t < 8; ++t) o[t] = zero8();
#pragma unroll
  for (int r = 0; r < 8; ++r) { mrow[r] = -3.0e38f; lrow[r] = 0.0f; }

  for (int c = 0; c < KVLEN / 128; ++c) {
    const int kvb = c * 128;
    __syncthreads();  // previous chunk's sK(P)/sVT consumers are done
#pragma unroll
    for (int i = 0; i < 8; ++i) {
      int task = tid + i * 256;   // 128 rows x 16 chunks of 8
      int row  = task >> 4;
      int c8   = (task & 15) << 3;
      cp16_g2l(Km + (size_t)(kvb + row) * HID + h * HD + c8,
               &sK[row * 128 + c8]);
      bf16x8 vv = *(const bf16x8*)(V + (size_t)(kvb + row) * HID + h * HD + c8);
#pragma unroll
      for (int j = 0; j < 8; ++j) sVT[(c8 + j) * 128 + row] = vv[j];
    }
    async_wait0();
    __syncthreads();

    // S = Q K^T
    v8f s[8];
#pragma unroll
    for (int t = 0; t < 8; ++t) {
      s[t] = zero8();
#pragma unroll
      for (int kf = 0; kf < 4; ++kf) {
        bf16x16 bf = ld_frag(&sK[(t * 16 + lm) * 128 + kf * 32], hi);
        s[t] = wmma_bf16(qf[kf], bf, s[t]);
      }
    }

    // online softmax
    float sc[8];
#pragma unroll
    for (int r = 0; r < 8; ++r) {
      float v = s[0][r];
#pragma unroll
      for (int t = 1; t < 8; ++t) v = fmaxf(v, s[t][r]);
#pragma unroll
      for (int off = 8; off; off >>= 1) v = fmaxf(v, __shfl_xor(v, off, 32));
      float mn = fmaxf(mrow[r], v);
      sc[r] = __expf(mrow[r] - mn);
      float rs = 0.0f;
#pragma unroll
      for (int t = 0; t < 8; ++t) {
        float p = __expf(s[t][r] - mn);
        s[t][r] = p;
        rs += p;
      }
#pragma unroll
      for (int off = 8; off; off >>= 1) rs += __shfl_xor(rs, off, 32);
      lrow[r] = lrow[r] * sc[r] + rs;
      mrow[r] = mn;
    }

    __syncthreads();  // all waves done reading sK -> safe to overwrite with P
#pragma unroll
    for (int t = 0; t < 8; ++t)
#pragma unroll
      for (int r = 0; r < 8; ++r)
        sP[(r + 8 * hi) * 128 + t * 16 + lm] = (bf16)s[t][r];

    // O = diag(sc) * O + P V
    bf16x16 pa[4];
#pragma unroll
    for (int kc = 0; kc < 4; ++kc)
      pa[kc] = ld_frag(&sP[lm * 128 + kc * 32], hi);
#pragma unroll
    for (int t = 0; t < 8; ++t) {
#pragma unroll
      for (int r = 0; r < 8; ++r) o[t][r] *= sc[r];
#pragma unroll
      for (int kc = 0; kc < 4; ++kc) {
        bf16x16 bv = ld_frag(&sVT[(t * 16 + lm) * 128 + kc * 32], hi);
        o[t] = wmma_bf16(pa[kc], bv, o[t]);
      }
    }
  }

#pragma unroll
  for (int t = 0; t < 8; ++t)
#pragma unroll
    for (int r = 0; r < 8; ++r) {
      float val = o[t][r] / lrow[r];
      O[(size_t)(q0 + r + 8 * hi) * HID + h * HD + t * 16 + lm] = (bf16)val;
    }
}

// ---------------------------------------------------------------------------
extern "C" void kernel_launch(void* const* d_in, const int* in_sizes, int n_in,
                              void* d_out, int out_size, void* d_ws, size_t ws_size,
                              hipStream_t stream) {
  const float* hs = (const float*)d_in[0];
  const float* kv = (const float*)d_in[1];
  const float* wq = (const float*)d_in[2];
  const float* wk = (const float*)d_in[3];
  const float* wv = (const float*)d_in[4];
  const float* wo = (const float*)d_in[5];
  const float* qw = (const float*)d_in[6];
  const float* kw = (const float*)d_in[7];
  (void)in_sizes; (void)n_in; (void)out_size; (void)ws_size;

  char* p = (char*)d_ws;
  float* f32s = (float*)p;  p += (size_t)KVLEN * HID * sizeof(float);  // proj scratch
  bf16*  Qn   = (bf16*)p;   p += (size_t)QLEN  * HID * sizeof(bf16);
  bf16*  Kn   = (bf16*)p;   p += (size_t)KVLEN * HID * sizeof(bf16);
  bf16*  Vb   = (bf16*)p;   p += (size_t)KVLEN * HID * sizeof(bf16);
  bf16*  Ab   = (bf16*)p;   // attention output bf16 [QLEN, HID]

  dim3 blk(256);
  const float inv_sqrt_d = 0.08838834764831845f;  // 1/sqrt(128)

  gemm_xwt<false, false><<<dim3(HID / 256, QLEN / 128), blk, 0, stream>>>(
      hs, wq, f32s, QLEN, HID, HID);
  rmsnorm_kernel<<<dim3(QLEN * NH / 8), blk, 0, stream>>>(f32s, qw, Qn, inv_sqrt_d);

  gemm_xwt<false, false><<<dim3(HID / 256, KVLEN / 128), blk, 0, stream>>>(
      kv, wk, f32s, KVLEN, HID, HID);
  rmsnorm_kernel<<<dim3(KVLEN * NH / 8), blk, 0, stream>>>(f32s, kw, Kn, 1.0f);

  gemm_xwt<false, true><<<dim3(HID / 256, KVLEN / 128), blk, 0, stream>>>(
      kv, wv, Vb, KVLEN, HID, HID);

  flash_attn_kernel<<<dim3(QLEN / 128, NH), blk, 0, stream>>>(Qn, Kn, Vb, Ab);

  gemm_xwt<true, false><<<dim3(HID / 256, QLEN / 128), blk, 0, stream>>>(
      Ab, wo, (float*)d_out, QLEN, HID, HID);
}
